// get_model_62380105007680
// MI455X (gfx1250) — compile-verified, matrix operand order checked by reading, not run on hardware
//
#include <hip/hip_runtime.h>
#include <hip/hip_bf16.h>
#include <math.h>

// ---------------------------------------------------------------------------
// Problem constants (match reference)
// ---------------------------------------------------------------------------
#define BB 16
#define NN 8192
#define GG 128
#define KK 32
#define CC 384

typedef _Float16 h8v  __attribute__((ext_vector_type(8)));
typedef _Float16 h16v __attribute__((ext_vector_type(16)));
typedef float    f8v  __attribute__((ext_vector_type(8)));

union H16U { h16v v; h8v h[2]; };

#define ROWS 520   // LDS activation row stride in halves (1040B: 16B aligned, bank-spread)

// ---------------------------------------------------------------------------
// A-fragment (16x32 f16) from LDS, row-major activation buffer.
// Lane l: row m = mtBase + (l&15); halves j<8 -> k = kc + hi*8 + j,
//         halves j>=8 -> k = kc + 16 + hi*8 + (j-8).   (ISA 7.12.2)
// ---------------------------------------------------------------------------
__device__ __forceinline__ h16v load_frag_act(const _Float16* base, int m, int kc, int hi) {
  H16U u;
  const _Float16* p = base + m * ROWS + kc + hi * 8;
  u.h[0] = *(const h8v*)p;
  u.h[1] = *(const h8v*)(p + 16);
  return u.v;
}

// B-fragment (32x16 f16): weight stored row-major [out(n)][in(k)] f16.
// Lane l holds column n; same K striping as A.
__device__ __forceinline__ h16v load_frag_w(const _Float16* w, int n, int ldk, int kc, int hi) {
  H16U u;
  const _Float16* p = w + (size_t)n * ldk + kc + hi * 8;
  u.h[0] = *(const h8v*)p;
  u.h[1] = *(const h8v*)(p + 16);
  return u.v;
}

// ---------------------------------------------------------------------------
// prep: fold BN, convert weights to f16 (L2-resident thereafter)
// ---------------------------------------------------------------------------
__global__ void prep_kernel(const float* __restrict__ w2, const float* __restrict__ w3,
                            const float* __restrict__ w4,
                            const float* __restrict__ g1, const float* __restrict__ bt1,
                            const float* __restrict__ m1, const float* __restrict__ v1,
                            const float* __restrict__ g3, const float* __restrict__ bt3,
                            const float* __restrict__ m3, const float* __restrict__ v3,
                            _Float16* __restrict__ w2h, _Float16* __restrict__ w3h,
                            _Float16* __restrict__ w4h,
                            float* __restrict__ bn1, float* __restrict__ bn3) {
  int i = blockIdx.x * blockDim.x + threadIdx.x;
  if (i < 256 * 128) w2h[i] = (_Float16)w2[i];
  if (i < 512 * 512) w3h[i] = (_Float16)w3[i];
  if (i < 384 * 512) w4h[i] = (_Float16)w4[i];
  if (i < 128) {
    float s = g1[i] * rsqrtf(v1[i] + 1e-5f);
    bn1[i] = s; bn1[128 + i] = bt1[i] - m1[i] * s;
  }
  if (i < 512) {
    float s = g3[i] * rsqrtf(v3[i] + 1e-5f);
    bn3[i] = s; bn3[512 + i] = bt3[i] - m3[i] * s;
  }
}

// ---------------------------------------------------------------------------
// FPS: 1 block per batch; xyz + dist fully LDS resident (128KB < 320KB/WGP)
// ---------------------------------------------------------------------------
#define FPS_T 1024
__global__ __launch_bounds__(FPS_T)
void fps_kernel(const float* __restrict__ xyz, float* __restrict__ centers) {
  extern __shared__ char sm[];
  float* sx = (float*)sm;
  float* sy = sx + NN;
  float* sz = sy + NN;
  float* sd = sz + NN;
  float* swv = sd + NN;              // 32
  int*   swi = (int*)(swv + 32);     // 32
  float* sPick = (float*)(swi + 32); // 4

  const int b = blockIdx.x, tid = threadIdx.x;
  const float* xb = xyz + (size_t)b * NN * 3;
  for (int i = tid; i < NN; i += FPS_T) {
    sx[i] = xb[i * 3 + 0]; sy[i] = xb[i * 3 + 1]; sz[i] = xb[i * 3 + 2];
  }
  __syncthreads();
  float px = sx[0], py = sy[0], pz = sz[0];
  if (tid == 0) {
    centers[(b * GG + 0) * 3 + 0] = px;
    centers[(b * GG + 0) * 3 + 1] = py;
    centers[(b * GG + 0) * 3 + 2] = pz;
  }
  for (int i = tid; i < NN; i += FPS_T) {
    float dx = sx[i] - px, dy = sy[i] - py, dz = sz[i] - pz;
    sd[i] = dx * dx + dy * dy + dz * dz;
  }
  __syncthreads();

  for (int it = 1; it < GG; ++it) {
    float bv = -1.0f; int bi = 0x7fffffff;
    for (int i = tid; i < NN; i += FPS_T) {
      float v = sd[i];
      if (v > bv) { bv = v; bi = i; }
    }
    #pragma unroll
    for (int off = 16; off > 0; off >>= 1) {
      float ov = __shfl_xor(bv, off, 32);
      int   oi = __shfl_xor(bi, off, 32);
      if (ov > bv || (ov == bv && oi < bi)) { bv = ov; bi = oi; }
    }
    int wid = tid >> 5;
    if ((tid & 31) == 0) { swv[wid] = bv; swi[wid] = bi; }
    __syncthreads();
    if (tid < 32) {
      bv = swv[tid]; bi = swi[tid];
      #pragma unroll
      for (int off = 16; off > 0; off >>= 1) {
        float ov = __shfl_xor(bv, off, 32);
        int   oi = __shfl_xor(bi, off, 32);
        if (ov > bv || (ov == bv && oi < bi)) { bv = ov; bi = oi; }
      }
      if (tid == 0) {
        float cx = sx[bi], cy = sy[bi], cz = sz[bi];
        sPick[0] = cx; sPick[1] = cy; sPick[2] = cz;
        centers[(b * GG + it) * 3 + 0] = cx;
        centers[(b * GG + it) * 3 + 1] = cy;
        centers[(b * GG + it) * 3 + 2] = cz;
      }
    }
    __syncthreads();
    px = sPick[0]; py = sPick[1]; pz = sPick[2];
    for (int i = tid; i < NN; i += FPS_T) {
      float dx = sx[i] - px, dy = sy[i] - py, dz = sz[i] - pz;
      float nd = dx * dx + dy * dy + dz * dz;
      if (nd < sd[i]) sd[i] = nd;
    }
    __syncthreads();
  }
}

// ---------------------------------------------------------------------------
// KNN: 1 block per (b,g); d^2 in LDS; 32 serial argmins (selection)
// ---------------------------------------------------------------------------
#define KNN_T 256
__global__ __launch_bounds__(KNN_T)
void knn_kernel(const float* __restrict__ xyz, const float* __restrict__ centers,
                float* __restrict__ neigh) {
  extern __shared__ char sm[];
  float* sd  = (float*)sm;          // 8192
  float* swv = sd + NN;             // 8
  int*   swi = (int*)(swv + 8);     // 8

  const int blk = blockIdx.x, tid = threadIdx.x;
  const int b = blk >> 7, g = blk & (GG - 1);
  const float cx = centers[(b * GG + g) * 3 + 0];
  const float cy = centers[(b * GG + g) * 3 + 1];
  const float cz = centers[(b * GG + g) * 3 + 2];
  const float* xb = xyz + (size_t)b * NN * 3;

  for (int i = tid; i < NN; i += KNN_T) {
    float dx = xb[i * 3 + 0] - cx, dy = xb[i * 3 + 1] - cy, dz = xb[i * 3 + 2] - cz;
    sd[i] = dx * dx + dy * dy + dz * dz;
  }
  __syncthreads();

  for (int k = 0; k < KK; ++k) {
    float bv = 3.402823466e+38f; int bi = 0x7fffffff;
    for (int i = tid; i < NN; i += KNN_T) {
      float v = sd[i];
      if (v < bv) { bv = v; bi = i; }
    }
    #pragma unroll
    for (int off = 16; off > 0; off >>= 1) {
      float ov = __shfl_xor(bv, off, 32);
      int   oi = __shfl_xor(bi, off, 32);
      if (ov < bv || (ov == bv && oi < bi)) { bv = ov; bi = oi; }
    }
    if ((tid & 31) == 0) { swv[tid >> 5] = bv; swi[tid >> 5] = bi; }
    __syncthreads();
    if (tid == 0) {
      float fv = swv[0]; int fi = swi[0];
      for (int w = 1; w < 8; ++w)
        if (swv[w] < fv || (swv[w] == fv && swi[w] < fi)) { fv = swv[w]; fi = swi[w]; }
      size_t o = (((size_t)(b * GG + g)) * KK + k) * 3;
      neigh[o + 0] = xb[fi * 3 + 0] - cx;
      neigh[o + 1] = xb[fi * 3 + 1] - cy;
      neigh[o + 2] = xb[fi * 3 + 2] - cz;
      sd[fi] = 3.402823466e+38f;
    }
    __syncthreads();
  }
}

// ---------------------------------------------------------------------------
// Morton (greedy NN tour): 1 block per batch, 128 threads
// ---------------------------------------------------------------------------
__global__ __launch_bounds__(128)
void morton_kernel(const float* __restrict__ centers, int* __restrict__ order) {
  __shared__ float cxs[GG], cys[GG], czs[GG];
  __shared__ int   vis[GG];
  __shared__ float swv[4];
  __shared__ int   swi[4];
  __shared__ int   sCur;
  const int b = blockIdx.x, tid = threadIdx.x;
  cxs[tid] = centers[(b * GG + tid) * 3 + 0];
  cys[tid] = centers[(b * GG + tid) * 3 + 1];
  czs[tid] = centers[(b * GG + tid) * 3 + 2];
  vis[tid] = (tid == 0) ? 1 : 0;
  if (tid == 0) order[b * GG + 0] = 0;
  __syncthreads();
  int cur = 0;
  for (int step = 1; step < GG; ++step) {
    float dx = cxs[tid] - cxs[cur], dy = cys[tid] - cys[cur], dz = czs[tid] - czs[cur];
    float bv = vis[tid] ? 3.402823466e+38f : (dx * dx + dy * dy + dz * dz);
    int bi = tid;
    #pragma unroll
    for (int off = 16; off > 0; off >>= 1) {
      float ov = __shfl_xor(bv, off, 32);
      int   oi = __shfl_xor(bi, off, 32);
      if (ov < bv || (ov == bv && oi < bi)) { bv = ov; bi = oi; }
    }
    if ((tid & 31) == 0) { swv[tid >> 5] = bv; swi[tid >> 5] = bi; }
    __syncthreads();
    if (tid == 0) {
      float fv = swv[0]; int fi = swi[0];
      for (int w = 1; w < 4; ++w)
        if (swv[w] < fv || (swv[w] == fv && swi[w] < fi)) { fv = swv[w]; fi = swi[w]; }
      vis[fi] = 1; sCur = fi; order[b * GG + step] = fi;
    }
    __syncthreads();
    cur = sCur;
    __syncthreads();
  }
}

// ---------------------------------------------------------------------------
// Encoder: 1 group per block, 256 threads (8 wave32), WMMA f16 GEMMs.
// Wave loop runs over N-tiles only; each B (weight) fragment is loaded once
// and reused for BOTH M-tiles (M=32 -> 2 tiles): 2 WMMAs per weight fetch,
// halving L2 weight traffic vs per-(mt,nt) tiling.
// ---------------------------------------------------------------------------
#define ENC_T 256
__global__ __launch_bounds__(ENC_T)
void encoder_kernel(const float* __restrict__ neigh, const float* __restrict__ centers,
                    const int* __restrict__ order,
                    const float* __restrict__ w1, const float* __restrict__ b1,
                    const float* __restrict__ bn1,
                    const _Float16* __restrict__ w2h, const float* __restrict__ b2,
                    const _Float16* __restrict__ w3h, const float* __restrict__ b3,
                    const float* __restrict__ bn3,
                    const _Float16* __restrict__ w4h, const float* __restrict__ b4,
                    float* __restrict__ out) {
  extern __shared__ char sm[];
  _Float16* sIn  = (_Float16*)sm;            // [32][ROWS]
  _Float16* sOut = sIn + 32 * ROWS;          // [32][ROWS]
  float* sFg = (float*)(sOut + 32 * ROWS);   // 256
  float* sPg = sFg + 256;                    // 96
  float* sC  = sPg + 96;                     // 4

  const int tid  = threadIdx.x;
  const int lane = tid & 31;
  const int wid  = tid >> 5;
  const int nl   = lane & 15;
  const int hi   = lane >> 4;

  const int blk = blockIdx.x;
  const int b = blk >> 7;
  const int i = blk & (GG - 1);
  const int g = order[b * GG + i];

  if (tid < KK * 3) sPg[tid] = neigh[((size_t)(b * GG + g)) * (KK * 3) + tid];
  if (tid < 3)      sC[tid]  = centers[(b * GG + g) * 3 + tid];
  __syncthreads();

  // Stage A: f1 = relu(bn1(pg @ w1^T + b1))  -> sIn[:, 0:128] f16
  for (int e = tid; e < 32 * 128; e += ENC_T) {
    int m = e >> 7, c = e & 127;
    float acc = b1[c]
              + sPg[m * 3 + 0] * w1[c * 3 + 0]
              + sPg[m * 3 + 1] * w1[c * 3 + 1]
              + sPg[m * 3 + 2] * w1[c * 3 + 2];
    acc = fmaxf(acc * bn1[c] + bn1[128 + c], 0.0f);
    sIn[m * ROWS + c] = (_Float16)acc;
  }
  __syncthreads();

  // Stage B: f = f1 @ w2^T + b2  [32,256] -> sIn[:, 256:512]
  for (int nt = wid; nt < 16; nt += 8) {
    f8v acc0 = {}, acc1 = {};
    #pragma unroll
    for (int kc = 0; kc < 128; kc += 32) {
      h16v a0 = load_frag_act(sIn, nl,      kc, hi);
      h16v a1 = load_frag_act(sIn, 16 + nl, kc, hi);
      h16v bw = load_frag_w(w2h, nt * 16 + nl, 128, kc, hi);
      acc0 = __builtin_amdgcn_wmma_f32_16x16x32_f16(false, a0, false, bw, (short)0, acc0, false, false);
      acc1 = __builtin_amdgcn_wmma_f32_16x16x32_f16(false, a1, false, bw, (short)0, acc1, false, false);
    }
    int n = nt * 16 + nl;
    float bb = b2[n];
    #pragma unroll
    for (int r = 0; r < 8; ++r) {
      int m = hi * 8 + r;
      sIn[m * ROWS + 256 + n]        = (_Float16)(acc0[r] + bb);
      sIn[(m + 16) * ROWS + 256 + n] = (_Float16)(acc1[r] + bb);
    }
  }
  __syncthreads();

  // fg = maxpool(f) over K -> broadcast into sIn[:, 0:256]
  if (tid < 256) {
    float mx = -3.402823466e+38f;
    for (int m = 0; m < 32; ++m) mx = fmaxf(mx, (float)sIn[m * ROWS + 256 + tid]);
    sFg[tid] = mx;
  }
  __syncthreads();
  for (int e = tid; e < 32 * 256; e += ENC_T) {
    int m = e >> 8, c = e & 255;
    sIn[m * ROWS + c] = (_Float16)sFg[c];
  }
  __syncthreads();

  // Stage C: hOut = relu(bn3(h @ w3^T + b3))  [32,512]
  for (int nt = wid; nt < 32; nt += 8) {
    f8v acc0 = {}, acc1 = {};
    #pragma unroll 4
    for (int kc = 0; kc < 512; kc += 32) {
      __builtin_prefetch(w3h + (size_t)(nt * 16 + nl) * 512 + kc + 32, 0, 0);
      h16v a0 = load_frag_act(sIn, nl,      kc, hi);
      h16v a1 = load_frag_act(sIn, 16 + nl, kc, hi);
      h16v bw = load_frag_w(w3h, nt * 16 + nl, 512, kc, hi);
      acc0 = __builtin_amdgcn_wmma_f32_16x16x32_f16(false, a0, false, bw, (short)0, acc0, false, false);
      acc1 = __builtin_amdgcn_wmma_f32_16x16x32_f16(false, a1, false, bw, (short)0, acc1, false, false);
    }
    int n = nt * 16 + nl;
    float bb = b3[n], sc = bn3[n], sh = bn3[512 + n];
    #pragma unroll
    for (int r = 0; r < 8; ++r) {
      int m = hi * 8 + r;
      float v0 = (acc0[r] + bb) * sc + sh;
      float v1 = (acc1[r] + bb) * sc + sh;
      sOut[m * ROWS + n]        = (_Float16)fmaxf(v0, 0.0f);
      sOut[(m + 16) * ROWS + n] = (_Float16)fmaxf(v1, 0.0f);
    }
  }
  __syncthreads();

  // Stage D: d = hOut @ w4^T  [32,384] -> sIn[:, 0:384] (bias folded after maxpool)
  for (int nt = wid; nt < 24; nt += 8) {
    f8v acc0 = {}, acc1 = {};
    #pragma unroll 4
    for (int kc = 0; kc < 512; kc += 32) {
      __builtin_prefetch(w4h + (size_t)(nt * 16 + nl) * 512 + kc + 32, 0, 0);
      h16v a0 = load_frag_act(sOut, nl,      kc, hi);
      h16v a1 = load_frag_act(sOut, 16 + nl, kc, hi);
      h16v bw = load_frag_w(w4h, nt * 16 + nl, 512, kc, hi);
      acc0 = __builtin_amdgcn_wmma_f32_16x16x32_f16(false, a0, false, bw, (short)0, acc0, false, false);
      acc1 = __builtin_amdgcn_wmma_f32_16x16x32_f16(false, a1, false, bw, (short)0, acc1, false, false);
    }
    int n = nt * 16 + nl;
    #pragma unroll
    for (int r = 0; r < 8; ++r) {
      int m = hi * 8 + r;
      sIn[m * ROWS + n]        = (_Float16)acc0[r];
      sIn[(m + 16) * ROWS + n] = (_Float16)acc1[r];
    }
  }
  __syncthreads();

  // token = maxpool + b4 ; add sin/cos positional embedding ; store
  for (int c = tid; c < CC; c += ENC_T) {
    float mx = -3.402823466e+38f;
    for (int m = 0; m < 32; ++m) mx = fmaxf(mx, (float)sIn[m * ROWS + c]);
    float token = mx + b4[c];
    int d = c >> 7, j2 = c & 127, j = j2 >> 1;
    float freq = __expf((float)j * (1.0f / 64.0f) * -9.210340371976184f); // 10000^{-j/64}
    float ang  = sC[d] * 6.283185307179586f * freq;
    float emb  = (j2 & 1) ? __cosf(ang) : __sinf(ang);
    out[(size_t)blk * CC + c] = token + emb;
  }
}

// ---------------------------------------------------------------------------
// Launch
// ---------------------------------------------------------------------------
extern "C" void kernel_launch(void* const* d_in, const int* in_sizes, int n_in,
                              void* d_out, int out_size, void* d_ws, size_t ws_size,
                              hipStream_t stream) {
  (void)in_sizes; (void)n_in; (void)out_size; (void)ws_size;
  const float* xyz = (const float*)d_in[0];
  const float* w1  = (const float*)d_in[1];
  const float* b1  = (const float*)d_in[2];
  const float* g1  = (const float*)d_in[3];
  const float* bt1 = (const float*)d_in[4];
  const float* m1  = (const float*)d_in[5];
  const float* v1  = (const float*)d_in[6];
  const float* w2  = (const float*)d_in[7];
  const float* b2  = (const float*)d_in[8];
  const float* w3  = (const float*)d_in[9];
  const float* b3  = (const float*)d_in[10];
  const float* g3  = (const float*)d_in[11];
  const float* bt3 = (const float*)d_in[12];
  const float* m3  = (const float*)d_in[13];
  const float* v3  = (const float*)d_in[14];
  const float* w4  = (const float*)d_in[15];
  const float* b4  = (const float*)d_in[16];
  float* out = (float*)d_out;

  char* ws = (char*)d_ws;
  float*    centers = (float*)(ws + 0);            // 24576 B
  int*      order   = (int*)  (ws + 24576);        //  8192 B
  float*    neigh   = (float*)(ws + 32768);        // 786432 B
  _Float16* w2h     = (_Float16*)(ws + 819200);    //  65536 B
  _Float16* w3h     = (_Float16*)(ws + 884736);    // 524288 B
  _Float16* w4h     = (_Float16*)(ws + 1409024);   // 393216 B
  float*    bn1     = (float*)(ws + 1802240);      //   1024 B
  float*    bn3     = (float*)(ws + 1803264);      //   4096 B

  prep_kernel<<<1024, 256, 0, stream>>>(w2, w3, w4, g1, bt1, m1, v1, g3, bt3, m3, v3,
                                        w2h, w3h, w4h, bn1, bn3);

  size_t fpsLds = (size_t)4 * NN * 4 + 32 * 4 + 32 * 4 + 16;
  fps_kernel<<<BB, FPS_T, fpsLds, stream>>>(xyz, centers);

  size_t knnLds = (size_t)NN * 4 + 8 * 4 + 8 * 4;
  knn_kernel<<<BB * GG, KNN_T, knnLds, stream>>>(xyz, centers, neigh);

  morton_kernel<<<BB, GG, 0, stream>>>(centers, order);

  size_t encLds = (size_t)2 * 32 * ROWS * 2 + 256 * 4 + 96 * 4 + 16;
  encoder_kernel<<<BB * GG, ENC_T, encLds, stream>>>(neigh, centers, order,
                                                     w1, b1, bn1, w2h, b2,
                                                     w3h, b3, bn3, w4h, b4, out);
}
